// GatedDeltaNet_85504208929344
// MI455X (gfx1250) — compile-verified, matrix operand order checked by reading, not run on hardware
//
#include <hip/hip_runtime.h>

// ---------------------------------------------------------------------------
// GatedDeltaNet forward for MI455X (gfx1250, wave32, WMMA + TDM)
// B=2 T=4096 D=1024 H=16 DK=64 DV=128 K=4
// ---------------------------------------------------------------------------

#define Bc 2
#define Tc 4096
#define Dc 1024
#define Hc 16
#define DKc 64
#define DVc 128
#define MTc (Bc * Tc)   // 8192
#define EPSc 1e-6f

typedef _Float16 half8  __attribute__((ext_vector_type(8)));
typedef _Float16 half16 __attribute__((ext_vector_type(16)));
typedef _Float16 half4  __attribute__((ext_vector_type(4)));
typedef float    floatx8 __attribute__((ext_vector_type(8)));
typedef unsigned int uint32x4 __attribute__((ext_vector_type(4)));
typedef int      int32x8 __attribute__((ext_vector_type(8)));
typedef int      int32x4 __attribute__((ext_vector_type(4)));

#if __has_builtin(__builtin_amdgcn_tensor_load_to_lds) && \
    __has_builtin(__builtin_amdgcn_s_wait_tensorcnt)
#define GDN_USE_TDM 1
#else
#define GDN_USE_TDM 0
#endif

// ---------------------------------------------------------------------------
// fp32 -> fp16 cast
// ---------------------------------------------------------------------------
__global__ __launch_bounds__(256)
void gdn_cast_f16(const float* __restrict__ src, _Float16* __restrict__ dst, int n) {
    int i = blockIdx.x * 256 + threadIdx.x;
    if (i < n) dst[i] = (_Float16)src[i];
}

#if GDN_USE_TDM
// ---------------------------------------------------------------------------
// Issue one 2-D TDM tile load: rows x 32 f16 tile from row-major global
// (row stride = strideElems f16) into LDS with padded rows (32 data + 8 pad
// f16 = row stride 40 f16).  D# packing per CDNA5 ISA ch.8.
// 6-arg builtin: (g0 u32x4, g1 i32x8, g2 i32x4, g3 i32x4, i32x8, cpol)
// ---------------------------------------------------------------------------
__device__ __attribute__((always_inline))
void gdn_tdm_load_tile(const _Float16* gsrc, const _Float16* ldst,
                       int rows, int strideElems) {
    const unsigned long long ga = (unsigned long long)(uintptr_t)gsrc;
    const unsigned int       la = (unsigned int)(uintptr_t)ldst;   // LDS byte addr

    uint32x4 g0;
    g0[0] = 1u;                                   // count=1, is_restore=0, gather=0
    g0[1] = la;                                   // lds_addr
    g0[2] = (unsigned int)ga;                     // global_addr[31:0]
    g0[3] = (unsigned int)((ga >> 32) & 0x1FFFFFFull) | (2u << 30);  // [56:32] | type=2

    int32x8 g1;
    // w0: workgroup_mask=0, data_size=1 (2B), pad_enable, pad_interval=3 (16 DW),
    //     pad_amount=3 (4 DW)  ->  LDS row stride = 64B data + 16B pad = 80B
    g1[0] = (int)((1u << 16) | (1u << 20) | (3u << 22) | (3u << 25));
    g1[1] = (int)(((unsigned)strideElems & 0xFFFFu) << 16);          // tensor_dim0[15:0]
    g1[2] = (int)((((unsigned)strideElems >> 16) & 0xFFFFu) |
                  (((unsigned)rows & 0xFFFFu) << 16));               // dim0[31:16] | dim1[15:0]
    g1[3] = (int)(((((unsigned)rows >> 16) & 0xFFFFu)) | (32u << 16)); // dim1[31:16] | tile_dim0=32
    g1[4] = (int)((unsigned)rows & 0xFFFFu);                         // tile_dim1=rows, tile_dim2=0
    g1[5] = (int)(unsigned)strideElems;                              // tensor_dim0_stride[31:0]
    g1[6] = 0;                                                       // stride0[47:32] | stride1[15:0]
    g1[7] = 0;

    int32x4 z4; z4[0] = 0; z4[1] = 0; z4[2] = 0; z4[3] = 0;
    int32x8 z8;
#pragma unroll
    for (int e = 0; e < 8; ++e) z8[e] = 0;
    __builtin_amdgcn_tensor_load_to_lds(g0, g1, z4, z4, z8, 0);
}
#endif

// ---------------------------------------------------------------------------
// WMMA GEMM: C[M,N] (f32) = A[M,K] (f16, row-major) * B[N,K]^T (f16, row-major)
// Block = 256 threads = 8 waves; block tile 128(M) x 64(N); K stepped by 32.
// TDM path: double-buffered LDS tiles filled by tensor_load_to_lds (wave 0),
// fragments read from LDS (ds_load_b128), layouts per ISA 7.12.2.
// ---------------------------------------------------------------------------
#define LDA 40   // padded LDS row stride in f16 (64B data + 16B pad)

__global__ __launch_bounds__(256)
void gdn_wmma_gemm_f16(const _Float16* __restrict__ A,
                       const _Float16* __restrict__ Bm,
                       float* __restrict__ C,
                       int M, int N, int K) {
    const int lane = threadIdx.x & 31;
    const int wave = threadIdx.x >> 5;
    const int m0   = (blockIdx.x * 8 + wave) * 16;
    const int n0   = blockIdx.y * 64;
    const int r    = lane & 15;    // row (A) / col (B) within 16
    const int hf   = lane >> 4;    // K-half selector

    floatx8 acc[4];
    acc[0] = (floatx8){0,0,0,0,0,0,0,0};
    acc[1] = acc[0]; acc[2] = acc[0]; acc[3] = acc[0];

#if GDN_USE_TDM
    __shared__ _Float16 As[2][128 * LDA];   // 2 x 10240 B
    __shared__ _Float16 Bs[2][64 * LDA];    // 2 x  5120 B

    const _Float16* Ablk = A  + (size_t)(blockIdx.x * 128) * K;
    const _Float16* Bblk = Bm + (size_t)(blockIdx.y * 64) * K;
    const int nk = K / 32;

    if (wave == 0) {
        gdn_tdm_load_tile(Ablk, &As[0][0], 128, K);
        gdn_tdm_load_tile(Bblk, &Bs[0][0], 64, K);
    }

    for (int i = 0; i < nk; ++i) {
        const int buf = i & 1;
        if (wave == 0) {
            if (i + 1 < nk) {
                gdn_tdm_load_tile(Ablk + (i + 1) * 32, &As[buf ^ 1][0], 128, K);
                gdn_tdm_load_tile(Bblk + (i + 1) * 32, &Bs[buf ^ 1][0], 64, K);
                __builtin_amdgcn_s_wait_tensorcnt(2);   // stage i complete
            } else {
                __builtin_amdgcn_s_wait_tensorcnt(0);
            }
        }
        __syncthreads();

        // A fragment from LDS: lane (r,hf) holds K = 8*hf+{0..7}, 16+8*hf+{0..7}
        half16 a;
        {
            const _Float16* Asr = &As[buf][(wave * 16 + r) * LDA];
            const half8 lo = *(const half8*)(Asr + 8 * hf);
            const half8 hi = *(const half8*)(Asr + 16 + 8 * hf);
#pragma unroll
            for (int e = 0; e < 8; ++e) { a[e] = lo[e]; a[e + 8] = hi[e]; }
        }
#pragma unroll
        for (int j = 0; j < 4; ++j) {
            const _Float16* Bsr = &Bs[buf][(j * 16 + r) * LDA];
            half16 b;
            {
                const half8 lo = *(const half8*)(Bsr + 8 * hf);
                const half8 hi = *(const half8*)(Bsr + 16 + 8 * hf);
#pragma unroll
                for (int e = 0; e < 8; ++e) { b[e] = lo[e]; b[e + 8] = hi[e]; }
            }
            acc[j] = __builtin_amdgcn_wmma_f32_16x16x32_f16(
                false, a, false, b, (short)0, acc[j], false, false);
        }
        __syncthreads();   // fragment reads done before next DMA overwrites
    }
#else
    // Fallback: direct-from-global operand fetch (round-1 body)
    const _Float16* arow = A + (size_t)(m0 + r) * K;
    for (int k0 = 0; k0 < K; k0 += 32) {
        if (k0 + 32 < K) __builtin_prefetch(arow + k0 + 32, 0, 1);
        half16 a;
        {
            const half8 lo = *(const half8*)(arow + k0 + 8 * hf);
            const half8 hi = *(const half8*)(arow + k0 + 16 + 8 * hf);
#pragma unroll
            for (int e = 0; e < 8; ++e) { a[e] = lo[e]; a[e + 8] = hi[e]; }
        }
#pragma unroll
        for (int j = 0; j < 4; ++j) {
            const _Float16* brow = Bm + (size_t)(n0 + j * 16 + r) * K;
            half16 b;
            {
                const half8 lo = *(const half8*)(brow + k0 + 8 * hf);
                const half8 hi = *(const half8*)(brow + k0 + 16 + 8 * hf);
#pragma unroll
                for (int e = 0; e < 8; ++e) { b[e] = lo[e]; b[e + 8] = hi[e]; }
            }
            acc[j] = __builtin_amdgcn_wmma_f32_16x16x32_f16(
                false, a, false, b, (short)0, acc[j], false, false);
        }
    }
#endif

    // C/D layout: lane L -> n = L&15 ; vgpr g -> m = g + 8*(L>>4)
#pragma unroll
    for (int j = 0; j < 4; ++j) {
#pragma unroll
        for (int g = 0; g < 8; ++g) {
            int m = m0 + g + 8 * hf;
            int n = n0 + j * 16 + r;
            C[(size_t)m * N + n] = acc[j][g];
        }
    }
}

// ---------------------------------------------------------------------------
// Gates: alpha = exp(-exp(A_log)*softplus(x@Wa^T + dt_bias)), beta = 2*sigmoid(x@Wb^T)
// ---------------------------------------------------------------------------
__global__ __launch_bounds__(256)
void gdn_gates(const float* __restrict__ x,
               const float* __restrict__ Wa, const float* __restrict__ Wb,
               const float* __restrict__ A_log, const float* __restrict__ dt_bias,
               float* __restrict__ alpha, float* __restrict__ beta) {
    const int bt   = blockIdx.x;
    const int wave = threadIdx.x >> 5;
    const int lane = threadIdx.x & 31;
    const float* xr = x + (size_t)bt * Dc;
    for (int s = 0; s < 4; ++s) {
        const int idx = wave * 4 + s;     // 0..15 alpha, 16..31 beta
        const int h   = idx & 15;
        const float* wr = (idx < 16 ? Wa : Wb) + (size_t)h * Dc;
        float p = 0.f;
        for (int i = lane; i < Dc; i += 32) p += xr[i] * wr[i];
#pragma unroll
        for (int m = 16; m; m >>= 1) p += __shfl_xor(p, m, 32);
        if (lane == 0) {
            if (idx < 16) {
                float z  = p + dt_bias[h];
                float sp = (z > 20.f) ? z : log1pf(expf(z));
                alpha[(size_t)bt * Hc + h] = expf(-expf(A_log[h]) * sp);
            } else {
                beta[(size_t)bt * Hc + h] = 2.f / (1.f + expf(-p));
            }
        }
    }
}

// ---------------------------------------------------------------------------
// Depthwise causal conv (K=4) + SiLU.  pre/out: (B,T,C) ; w: (C,4)
// ---------------------------------------------------------------------------
__global__ __launch_bounds__(256)
void gdn_conv_silu(const float* __restrict__ pre, const float* __restrict__ w,
                   float* __restrict__ out, int C, int total) {
    int i = blockIdx.x * 256 + threadIdx.x;
    if (i >= total) return;
    int c  = i % C;
    int bt = i / C;
    int t  = bt % Tc;
    float y = 0.f;
#pragma unroll
    for (int j = 0; j < 4; ++j) {
        int tt = t - 3 + j;
        if (tt >= 0) y += w[c * 4 + j] * pre[(size_t)(bt - (3 - j)) * C + c];
    }
    out[i] = y / (1.f + expf(-y));   // silu
}

// ---------------------------------------------------------------------------
// Per-head L2 norm over 64 elements, in place.  One wave per row.
// ---------------------------------------------------------------------------
__global__ __launch_bounds__(256)
void gdn_l2norm64(float* __restrict__ q, int nrows) {
    int row  = blockIdx.x * 8 + (threadIdx.x >> 5);
    int lane = threadIdx.x & 31;
    if (row >= nrows) return;
    float* r = q + (size_t)row * 64;
    float2 v = *(float2*)(r + lane * 2);
    float ss = v.x * v.x + v.y * v.y;
#pragma unroll
    for (int m = 16; m; m >>= 1) ss += __shfl_xor(ss, m, 32);
    float sc = 1.f / fmaxf(sqrtf(ss), EPSc);
    v.x *= sc; v.y *= sc;
    *(float2*)(r + lane * 2) = v;
}

// ---------------------------------------------------------------------------
// Gated delta-rule recurrence.  One block per (b,h) chain, 256 threads.
// S (64 x 128) fp32 in registers: thread (col = tid&127, rh = tid>>7) owns
// S[rh*32 .. rh*32+31][col].
// ---------------------------------------------------------------------------
__global__ __launch_bounds__(256)
void gdn_recurrence(const float* __restrict__ q, const float* __restrict__ k,
                    const float* __restrict__ v, const float* __restrict__ alpha,
                    const float* __restrict__ beta, float* __restrict__ o) {
    const int bh  = blockIdx.x;          // 0..31
    const int b   = bh >> 4;
    const int h   = bh & 15;
    const int tid = threadIdx.x;
    const int col = tid & 127;
    const int rh  = tid >> 7;            // 0 or 1

    __shared__ float sk[64], sq[64], sv[128], red[256], sab[2];

    float S[32];
#pragma unroll
    for (int i = 0; i < 32; ++i) S[i] = 0.f;

    for (int t = 0; t < Tc; ++t) {
        const size_t bt  = (size_t)b * Tc + t;
        const size_t qko = bt * (Hc * DKc) + (size_t)h * DKc;
        const size_t vo  = bt * (Hc * DVc) + (size_t)h * DVc;

        if (tid < 64)        sk[tid]       = k[qko + tid];
        else if (tid < 128)  sq[tid - 64]  = q[qko + tid - 64];
        else                 sv[tid - 128] = v[vo + tid - 128];
        if (tid == 0) sab[0] = alpha[bt * Hc + h];
        if (tid == 1) sab[1] = beta[bt * Hc + h];
        __syncthreads();

        float p = 0.f;
#pragma unroll
        for (int i = 0; i < 32; ++i) p += sk[rh * 32 + i] * S[i];
        red[tid] = p;
        __syncthreads();

        const float kS = red[col] + red[128 + col];
        const float a  = sab[0];
        const float bb = sab[1];
        const float vv = sv[col];
        __syncthreads();

        float po = 0.f;
#pragma unroll
        for (int i = 0; i < 32; ++i) {
            const float kd = sk[rh * 32 + i];
            const float s  = a * (S[i] - bb * kd * kS) + kd * vv;
            S[i] = s;
            po += s * sq[rh * 32 + i];
        }
        red[tid] = po;
        __syncthreads();

        if (tid < 128) o[vo + tid] = red[tid] + red[tid + 128];
        __syncthreads();
    }
}

// ---------------------------------------------------------------------------
// Per-head RMS norm over 128 elems + cast to fp16.  One wave per row.
// ---------------------------------------------------------------------------
__global__ __launch_bounds__(256)
void gdn_rmsnorm128_cast(const float* __restrict__ o, _Float16* __restrict__ oh,
                         int nrows) {
    int row  = blockIdx.x * 8 + (threadIdx.x >> 5);
    int lane = threadIdx.x & 31;
    if (row >= nrows) return;
    const float* r = o + (size_t)row * 128;
    float4 v = *(const float4*)(r + lane * 4);
    float ss = v.x * v.x + v.y * v.y + v.z * v.z + v.w * v.w;
#pragma unroll
    for (int m = 16; m; m >>= 1) ss += __shfl_xor(ss, m, 32);
    float sc = rsqrtf(ss * (1.f / 128.f) + EPSc);
    half4 hv;
    hv[0] = (_Float16)(v.x * sc);
    hv[1] = (_Float16)(v.y * sc);
    hv[2] = (_Float16)(v.z * sc);
    hv[3] = (_Float16)(v.w * sc);
    *(half4*)(oh + (size_t)row * 128 + lane * 4) = hv;
}

// ---------------------------------------------------------------------------
// Host launcher
// ---------------------------------------------------------------------------
extern "C" void kernel_launch(void* const* d_in, const int* in_sizes, int n_in,
                              void* d_out, int out_size, void* d_ws, size_t ws_size,
                              hipStream_t stream) {
    (void)in_sizes; (void)n_in; (void)out_size; (void)ws_size;

    // setup_inputs order: x, Wq, Wk, Wv, Wb, Wa, A_log, dt_bias, Wo, conv_q, conv_k, conv_v
    const float* x       = (const float*)d_in[0];
    const float* Wq      = (const float*)d_in[1];
    const float* Wk      = (const float*)d_in[2];
    const float* Wv      = (const float*)d_in[3];
    const float* Wb      = (const float*)d_in[4];
    const float* Wa      = (const float*)d_in[5];
    const float* A_log   = (const float*)d_in[6];
    const float* dt_bias = (const float*)d_in[7];
    const float* Wo      = (const float*)d_in[8];
    const float* conv_q  = (const float*)d_in[9];
    const float* conv_k  = (const float*)d_in[10];
    const float* conv_v  = (const float*)d_in[11];
    float* out = (float*)d_out;

    char* ws = (char*)d_ws;
    const size_t MiB = (size_t)1 << 20;
    _Float16* xh  = (_Float16*)(ws + 0 * MiB);    // 16 MiB  (8192x1024)
    _Float16* Wqh = (_Float16*)(ws + 16 * MiB);   //  2 MiB
    _Float16* Wkh = (_Float16*)(ws + 18 * MiB);   //  2 MiB
    _Float16* Wvh = (_Float16*)(ws + 20 * MiB);   //  4 MiB
    _Float16* Woh = (_Float16*)(ws + 24 * MiB);   //  4 MiB
    float* qpre = (float*)(ws + 28 * MiB);        // 32 MiB
    float* kpre = (float*)(ws + 60 * MiB);        // 32 MiB
    float* obuf = (float*)(ws + 28 * MiB);        // 64 MiB (aliases qpre+kpre)
    float* vpre = (float*)(ws + 92 * MiB);        // 64 MiB
    _Float16* oh = (_Float16*)(ws + 92 * MiB);    // 32 MiB (aliases vpre)
    float* qbuf = (float*)(ws + 156 * MiB);       // 32 MiB
    float* kbuf = (float*)(ws + 188 * MiB);       // 32 MiB
    float* vbuf = (float*)(ws + 220 * MiB);       // 64 MiB
    float* alpha = (float*)(ws + 284 * MiB);      // 0.5 MiB
    float* beta  = (float*)(ws + 284 * MiB + 512 * 1024);

    const int M = MTc;  // 8192

    // 1) cast to fp16
    {
        int n;
        n = MTc * Dc;        gdn_cast_f16<<<(n + 255) / 256, 256, 0, stream>>>(x,  xh,  n);
        n = (Hc * DKc) * Dc; gdn_cast_f16<<<(n + 255) / 256, 256, 0, stream>>>(Wq, Wqh, n);
        n = (Hc * DKc) * Dc; gdn_cast_f16<<<(n + 255) / 256, 256, 0, stream>>>(Wk, Wkh, n);
        n = (Hc * DVc) * Dc; gdn_cast_f16<<<(n + 255) / 256, 256, 0, stream>>>(Wv, Wvh, n);
        n = Dc * (Hc * DVc); gdn_cast_f16<<<(n + 255) / 256, 256, 0, stream>>>(Wo, Woh, n);
    }

    // 2) projection GEMMs (WMMA + TDM)
    gdn_wmma_gemm_f16<<<dim3(M / 128, (Hc * DKc) / 64), 256, 0, stream>>>(
        xh, Wqh, qpre, M, Hc * DKc, Dc);
    gdn_wmma_gemm_f16<<<dim3(M / 128, (Hc * DKc) / 64), 256, 0, stream>>>(
        xh, Wkh, kpre, M, Hc * DKc, Dc);
    gdn_wmma_gemm_f16<<<dim3(M / 128, (Hc * DVc) / 64), 256, 0, stream>>>(
        xh, Wvh, vpre, M, Hc * DVc, Dc);

    // 3) gates (fp32)
    gdn_gates<<<M, 256, 0, stream>>>(x, Wa, Wb, A_log, dt_bias, alpha, beta);

    // 4) depthwise causal conv + SiLU
    {
        int total;
        total = M * (Hc * DKc);
        gdn_conv_silu<<<(total + 255) / 256, 256, 0, stream>>>(qpre, conv_q, qbuf, Hc * DKc, total);
        gdn_conv_silu<<<(total + 255) / 256, 256, 0, stream>>>(kpre, conv_k, kbuf, Hc * DKc, total);
        total = M * (Hc * DVc);
        gdn_conv_silu<<<(total + 255) / 256, 256, 0, stream>>>(vpre, conv_v, vbuf, Hc * DVc, total);
    }

    // 5) per-head L2 norm on q, k
    {
        int nrows = M * Hc;
        gdn_l2norm64<<<(nrows + 7) / 8, 256, 0, stream>>>(qbuf, nrows);
        gdn_l2norm64<<<(nrows + 7) / 8, 256, 0, stream>>>(kbuf, nrows);
    }

    // 6) sequential gated delta-rule scan (32 chains)
    gdn_recurrence<<<Bc * Hc, 256, 0, stream>>>(qbuf, kbuf, vbuf, alpha, beta, obuf);

    // 7) RMS norm + cast fp16
    {
        int nrows = M * Hc;
        gdn_rmsnorm128_cast<<<(nrows + 7) / 8, 256, 0, stream>>>(obuf, oh, nrows);
    }

    // 8) output projection GEMM (WMMA) -> d_out (fp32)
    gdn_wmma_gemm_f16<<<dim3(M / 128, Dc / 64), 256, 0, stream>>>(
        oh, Woh, out, M, Dc, Hc * DVc);
}
